// SelfAttention_18794776887430
// MI455X (gfx1250) — matrix operand order repaired and checked by reading, not compiled, not run on hardware
//
#include <hip/hip_runtime.h>
#include <hip/hip_bf16.h>

typedef __bf16 bf16_t;
typedef __attribute__((ext_vector_type(16))) __bf16 v16bf;
typedef __attribute__((ext_vector_type(8)))  float  v8f;

#define B_   2
#define H_   16
#define T_   2048
#define E_   2048
#define HD_  128
#define SCALE_ 0.08838834764831845f   // 1/sqrt(128)

union FragU { v16bf v; uint4 u[2]; };

static __device__ __forceinline__ v8f wmma_bf16(v16bf a, v16bf b, v8f c) {
  // D = A(16x32 bf16) * B(32x16 bf16) + C(16x16 f32)
  return __builtin_amdgcn_wmma_f32_16x16x32_bf16(false, a, false, b, (short)0, c, false, false);
}

// A-style fragment: two 8-element (16B) chunks per lane
static __device__ __forceinline__ v16bf ldfrag2x8(const bf16_t* p, int off0, int off1) {
  FragU f;
  f.u[0] = *(const uint4*)(p + off0);
  f.u[1] = *(const uint4*)(p + off1);
  return f.v;
}
// B-style fragment: 16 contiguous elements (32B) per lane
static __device__ __forceinline__ v16bf ldfrag16(const bf16_t* p) {
  FragU f;
  f.u[0] = ((const uint4*)p)[0];
  f.u[1] = ((const uint4*)p)[1];
  return f.v;
}

// ---------------------------------------------------------------------------
// Tiled bf16 WMMA GEMM: C[M,N] = A[M,K] * B[K,N]
//   A_F32:   A source is f32 (convert to bf16 in LDS) else bf16
//   SCATTER: epilogue scatters bf16 into Q/K/V [B,H,T,hd]; else f32 store to C
// Tiles: BM=128, BN=128, BK=32; 8 waves as 4(M)x2(N); 32x64 per wave.
// LDS double-buffered; next tile staged through registers while computing.
// ---------------------------------------------------------------------------
template<bool A_F32, bool SCATTER>
__global__ __launch_bounds__(256) void gemm_wmma_128(
    const void* __restrict__ Ap, const float* __restrict__ Bp,
    float* __restrict__ Cout,
    bf16_t* __restrict__ Qb, bf16_t* __restrict__ Kb, bf16_t* __restrict__ Vb,
    int M, int N, int K)
{
  __shared__ __align__(16) bf16_t As[2][128 * 32];   // [row][k]
  __shared__ __align__(16) bf16_t Bs[2][128 * 32];   // transposed: [n][k]
  const int tid  = threadIdx.x;
  const int lane = tid & 31, w = tid >> 5;
  const int wm = w >> 1, wn = w & 1;
  const int hi = lane >> 4, ln = lane & 15;
  const int mbase = blockIdx.y * 128, nbase = blockIdx.x * 128;

  v8f acc[2][4];
  v8f zf = {};
  #pragma unroll
  for (int fm = 0; fm < 2; ++fm)
    #pragma unroll
    for (int fn = 0; fn < 4; ++fn) acc[fm][fn] = zf;

  float ar[16], br[16];
  auto load_tile = [&](int k0) {
    #pragma unroll
    for (int i = 0; i < 16; ++i) {
      int e = i * 256 + tid; int r = e >> 5; int c = e & 31;
      if (A_F32) ar[i] = ((const float*)Ap)[(size_t)(mbase + r) * K + k0 + c];
      else       ar[i] = (float)((const bf16_t*)Ap)[(size_t)(mbase + r) * K + k0 + c];
    }
    #pragma unroll
    for (int i = 0; i < 16; ++i) {
      int e = i * 256 + tid; int kk = e >> 7; int n = e & 127;
      br[i] = Bp[(size_t)(k0 + kk) * N + nbase + n];
    }
  };
  auto store_tile = [&](int buf) {
    #pragma unroll
    for (int i = 0; i < 16; ++i) {
      int e = i * 256 + tid;
      As[buf][e] = (bf16_t)ar[i];
    }
    #pragma unroll
    for (int i = 0; i < 16; ++i) {
      int e = i * 256 + tid; int kk = e >> 7; int n = e & 127;
      Bs[buf][n * 32 + kk] = (bf16_t)br[i];
    }
  };

  load_tile(0);
  int cur = 0;
  for (int k0 = 0; k0 < K; k0 += 32) {
    store_tile(cur);
    __syncthreads();
    if (k0 + 32 < K) load_tile(k0 + 32);   // overlap next global fetch w/ WMMA

    v16bf af[2];
    #pragma unroll
    for (int fm = 0; fm < 2; ++fm) {
      int row = wm * 32 + fm * 16 + ln;
      int cb = hi ? 16 : 0;                                  // contiguous k = 16*hi + s
      af[fm] = ldfrag2x8(As[cur], row * 32 + cb, row * 32 + cb + 8);
    }
    #pragma unroll
    for (int fn = 0; fn < 4; ++fn) {
      int n = wn * 64 + fn * 16 + ln;
      v16bf bfg = ldfrag16(&Bs[cur][n * 32 + (hi ? 16 : 0)]); // contiguous k = 16*hi + s, same as A
      #pragma unroll
      for (int fm = 0; fm < 2; ++fm)
        acc[fm][fn] = wmma_bf16(af[fm], bfg, acc[fm][fn]);
    }
    cur ^= 1;
  }

  // epilogue; C layout: elem i -> row = i + 8*hi, col = ln
  #pragma unroll
  for (int fm = 0; fm < 2; ++fm)
    #pragma unroll
    for (int fn = 0; fn < 4; ++fn)
      #pragma unroll
      for (int i = 0; i < 8; ++i) {
        int row = mbase + wm * 32 + fm * 16 + i + hi * 8;
        int col = nbase + wn * 64 + fn * 16 + ln;
        float v = acc[fm][fn][i];
        if (SCATTER) {
          int which = col >> 11, e = col & 2047;             // Q/K/V split over 3E
          int h = e >> 7, d = e & 127;
          int b = row >> 11, t = row & 2047;                 // row over B*T
          size_t dst = (((size_t)b * H_ + h) * T_ + t) * HD_ + d;
          bf16_t bv = (bf16_t)v;
          if (which == 0) Qb[dst] = bv;
          else if (which == 1) Kb[dst] = bv;
          else Vb[dst] = bv;
        } else {
          Cout[(size_t)row * N + col] = v;
        }
      }
}

// ---------------------------------------------------------------------------
// RoPE in-place on Q and K ([B,H,T,128] bf16). One thread per (tensor, b,h,t,d<64).
// ---------------------------------------------------------------------------
__global__ __launch_bounds__(256) void rope_inplace(bf16_t* __restrict__ Qb,
                                                    bf16_t* __restrict__ Kb)
{
  const int PT = B_ * H_ * T_ * 64;                 // pairs per tensor
  int idx = blockIdx.x * 256 + threadIdx.x;
  int which = (idx >= PT) ? 1 : 0;
  int r = idx - which * PT;
  int bh = r >> 17;                                 // T_*64 = 2^17
  int t  = (r >> 6) & (T_ - 1);
  int d  = r & 63;
  bf16_t* p = (which ? Kb : Qb) + ((size_t)bh * T_ + t) * HD_ + d;
  float x1 = (float)p[0], x2 = (float)p[64];
  // inv_freq = 10000^(-d/64) = exp(-d * ln(1e4)/64)
  float ang = (float)t * __expf(-(float)d * 0.14391156831212787f);
  float s, c;
  __sincosf(ang, &s, &c);
  p[0]  = (bf16_t)(x1 * c - x2 * s);
  p[64] = (bf16_t)(x2 * c + x1 * s);
}

// ---------------------------------------------------------------------------
// Flash attention. Grid (qt=T/128, H, B), 256 threads (8 waves).
// Wave w owns q-rows [qt*128 + 16w, +16). Key tiles of 64 with causal cutoff.
// K tile staged with GLOBAL_LOAD_ASYNC_TO_LDS_B128 (ASYNCcnt path).
// ---------------------------------------------------------------------------
__global__ __launch_bounds__(256) void attn_fwd(
    const bf16_t* __restrict__ Qb, const bf16_t* __restrict__ Kb,
    const bf16_t* __restrict__ Vb, const int* __restrict__ amask,
    bf16_t* __restrict__ Ab)
{
  __shared__ __align__(16) bf16_t Ks[64 * 128];     // [key][d]   (QK^T B-frags)
  __shared__ __align__(16) bf16_t Vt[128 * 64];     // [d][key]   (PV   B-frags)
  __shared__ __align__(16) bf16_t Pb[8 * 16 * 64];  // per-wave P scratch
  const int tid = threadIdx.x, lane = tid & 31, w = tid >> 5;
  const int hi = lane >> 4, ln = lane & 15;
  const int qt = blockIdx.x, h = blockIdx.y, b = blockIdx.z;
  const int qr0 = qt * 128 + w * 16;
  const size_t bh = (size_t)b * H_ + h;

  // LDS byte address of Ks (flat LDS pointer truncates to LDS offset)
  const unsigned int ks_lds = (unsigned int)(size_t)(&Ks[0]);

  // Q fragments (A layout), kept in registers: 16 rows x 128 d = 4 frags
  const bf16_t* Qrow = Qb + (bh * T_ + qr0 + ln) * HD_;
  v16bf aq[4];
  #pragma unroll
  for (int c = 0; c < 4; ++c) {
    int cb = c * 32 + (hi ? 16 : 0);
    aq[c] = ldfrag2x8(Qrow, cb, cb + 8);
  }

  int qm[8];
  float mrun[8], lrun[8];
  v8f o[8];
  v8f zf = {};
  #pragma unroll
  for (int i = 0; i < 8; ++i) { mrun[i] = -1e30f; lrun[i] = 0.f; }
  #pragma unroll
  for (int f = 0; f < 8; ++f) o[f] = zf;
  #pragma unroll
  for (int i = 0; i < 8; ++i) qm[i] = amask[b * T_ + qr0 + i + hi * 8];

  bf16_t* Pw = Pb + w * (16 * 64);
  const int numkt = qt * 2 + 2;                     // causal: keys <= qtile max

  for (int kt = 0; kt < numkt; ++kt) {
    const int kb = kt * 64;

    // --- async stage K tile: contiguous 16KB global -> LDS, 4x16B per lane
    {
      unsigned long long gsrc =
          (unsigned long long)(size_t)(Kb + (bh * T_ + kb) * HD_);
      #pragma unroll
      for (int i = 0; i < 4; ++i) {
        unsigned int chunk = (unsigned int)(i * 256 + tid) * 16u;
        unsigned int laddr = ks_lds + chunk;
        unsigned long long gaddr = gsrc + chunk;
        asm volatile("global_load_async_to_lds_b128 %0, %1, off"
                     :: "v"(laddr), "v"(gaddr) : "memory");
      }
    }
    // --- stage V transposed through VGPRs
    const bf16_t* Vsrc = Vb + (bh * T_ + kb) * HD_;
    #pragma unroll
    for (int i = 0; i < 32; ++i) {
      int e = i * 256 + tid; int key = e >> 7; int d = e & 127;
      Vt[d * 64 + key] = Vsrc[e];
    }
    asm volatile("s_wait_asynccnt 0x0" ::: "memory");
    __syncthreads();

    // S = Q K^T over 4 key-column fragments, K-dim 128 = 4 WMMA steps
    v8f sf[4];
    #pragma unroll
    for (int n = 0; n < 4; ++n) {
      sf[n] = zf;
      #pragma unroll
      for (int c = 0; c < 4; ++c) {
        v16bf bfg = ldfrag16(&Ks[(n * 16 + ln) * 128 + c * 32 + (hi ? 16 : 0)]);
        sf[n] = wmma_bf16(aq[c], bfg, sf[n]);
      }
    }

    // scale + masks (reference semantics: masked -> -1e9 before softmax)
    int km[4];
    #pragma unroll
    for (int n = 0; n < 4; ++n) km[n] = amask[b * T_ + kb + n * 16 + ln];
    #pragma unroll
    for (int n = 0; n < 4; ++n)
      #pragma unroll
      for (int i = 0; i < 8; ++i) {
        int key = kb + n * 16 + ln;
        int q = qr0 + i + hi * 8;
        float s = sf[n][i] * SCALE_;
        if (key > q || km[n] == 0 || qm[i] == 0) s = -1e9f;
        sf[n][i] = s;
      }

    // online softmax per q-row (16-lane butterfly within half-wave group)
    #pragma unroll
    for (int i = 0; i < 8; ++i) {
      float mt = fmaxf(fmaxf(sf[0][i], sf[1][i]), fmaxf(sf[2][i], sf[3][i]));
      #pragma unroll
      for (int off = 1; off < 16; off <<= 1) mt = fmaxf(mt, __shfl_xor(mt, off, 32));
      float mnew  = fmaxf(mrun[i], mt);
      float alpha = __expf(mrun[i] - mnew);
      mrun[i] = mnew;
      float rs = 0.f;
      #pragma unroll
      for (int n = 0; n < 4; ++n) {
        float p = __expf(sf[n][i] - mnew);
        sf[n][i] = p;
        rs += p;
      }
      #pragma unroll
      for (int off = 1; off < 16; off <<= 1) rs += __shfl_xor(rs, off, 32);
      lrun[i] = lrun[i] * alpha + rs;
      #pragma unroll
      for (int f = 0; f < 8; ++f) o[f][i] *= alpha;
    }

    // P (C layout) -> per-wave LDS -> A layout fragments
    #pragma unroll
    for (int n = 0; n < 4; ++n)
      #pragma unroll
      for (int i = 0; i < 8; ++i)
        Pw[(i + hi * 8) * 64 + n * 16 + ln] = (bf16_t)sf[n][i];
    asm volatile("s_wait_dscnt 0" ::: "memory");

    // O += P @ V  (8 d-frags x 2 key-chunks of 32)
    #pragma unroll
    for (int f = 0; f < 8; ++f) {
      #pragma unroll
      for (int kc = 0; kc < 64; kc += 32) {
        int cb = kc + (hi ? 16 : 0);
        v16bf pa  = ldfrag2x8(Pw + ln * 64, cb, cb + 8);
        v16bf vbf = ldfrag16(&Vt[(f * 16 + ln) * 64 + kc + (hi ? 16 : 0)]);
        o[f] = wmma_bf16(pa, vbf, o[f]);
      }
    }
    __syncthreads();
  }

  // normalize + store [B,T,E] bf16 (e = h*128 + d)
  #pragma unroll
  for (int i = 0; i < 8; ++i) {
    float linv = 1.f / lrun[i];
    int q = qr0 + i + hi * 8;
    size_t base = ((size_t)b * T_ + q) * E_ + h * HD_;
    #pragma unroll
    for (int f = 0; f < 8; ++f)
      Ab[base + f * 16 + ln] = (bf16_t)(o[f][i] * linv);
  }
}

// ---------------------------------------------------------------------------
extern "C" void kernel_launch(void* const* d_in, const int* in_sizes, int n_in,
                              void* d_out, int out_size, void* d_ws, size_t ws_size,
                              hipStream_t stream) {
  (void)in_sizes; (void)n_in; (void)out_size; (void)ws_size;
  const float* x     = (const float*)d_in[0];
  const int*   amask = (const int*)d_in[1];
  const float* Wqkv  = (const float*)d_in[2];
  const float* Wproj = (const float*)d_in[3];

  const size_t SZ = (size_t)B_ * H_ * T_ * HD_ * sizeof(bf16_t);   // 16 MB
  char* wsp = (char*)d_ws;
  bf16_t* Qb = (bf16_t*)(wsp);
  bf16_t* Kb = (bf16_t*)(wsp + SZ);
  bf16_t* Vb = (bf16_t*)(wsp + 2 * SZ);
  bf16_t* Ab = (bf16_t*)(wsp + 3 * SZ);

  // 1) QKV = x @ W_qkv, scatter bf16 Q/K/V into [B,H,T,hd]
  gemm_wmma_128<true, true><<<dim3(48, 32), 256, 0, stream>>>(
      x, Wqkv, nullptr, Qb, Kb, Vb, B_ * T_, 3 * E_, E_);
  // 2) RoPE on Q and K
  rope_inplace<<<(2 * B_ * H_ * T_ * 64) / 256, 256, 0, stream>>>(Qb, Kb);
  // 3) flash attention -> Ab [B,T,E] bf16
  attn_fwd<<<dim3(T_ / 128, H_, B_), 256, 0, stream>>>(Qb, Kb, Vb, amask, Ab);
  // 4) out = Ab @ W_proj (f32 out)
  gemm_wmma_128<false, false><<<dim3(16, 32), 256, 0, stream>>>(
      Ab, Wproj, (float*)d_out, nullptr, nullptr, nullptr, B_ * T_, E_, E_);
}
